// HiDRA_13683765805502
// MI455X (gfx1250) — matrix operand order repaired and checked by reading, not compiled, not run on hardware
//
#include <hip/hip_runtime.h>
#include <hip/hip_bf16.h>
#include <math.h>

// ---------------------------------------------------------------------------
// HiDRA forward, CDNA5 (gfx1250), wave32.
// v_wmma_f32_16x16x32_f16 + global_load_async_to_lds_* + b128 LDS fragments.
// B=2048, P=332, M=64, G=15000, D=512, D1G=17, D1P=21
// ---------------------------------------------------------------------------

typedef __attribute__((ext_vector_type(16))) _Float16 v16h;
typedef __attribute__((ext_vector_type(8)))  _Float16 h8;
typedef __attribute__((ext_vector_type(4)))  _Float16 h4;
typedef __attribute__((ext_vector_type(8)))  float    v8f;
typedef __attribute__((ext_vector_type(4)))  float    f4;

union V16 { v16h v; h8 h[2]; };
union FAcc { v8f v; float f[8]; };

#define BATCH_N 2048
#define NGENES  15000

// --- CDNA5 async global->LDS helpers (inline asm; tracked by ASYNCcnt) -----
__device__ __forceinline__ unsigned lds_off_of(const void* p) {
    return (unsigned)(uintptr_t)p;   // generic LDS addr: low 32 bits = offset
}
__device__ __forceinline__ void async_load_lds_b32(unsigned lds_off, const void* g) {
    asm volatile("global_load_async_to_lds_b32 %0, %1, off"
                 :: "v"(lds_off), "v"((unsigned long long)(uintptr_t)g) : "memory");
}
__device__ __forceinline__ void async_load_lds_b128(unsigned lds_off, const void* g) {
    asm volatile("global_load_async_to_lds_b128 %0, %1, off"
                 :: "v"(lds_off), "v"((unsigned long long)(uintptr_t)g) : "memory");
}
__device__ __forceinline__ void wait_async0() {
    asm volatile("s_wait_asynccnt 0x0" ::: "memory");
}

// ---------------------------------------------------------------------------
// Weight pre-pass: W [batch][K][N] f32  ->  WT [batch][Np][Kp] f16,
// transposed + zero-padded (Kp%32==0, Np%16==0).
// ---------------------------------------------------------------------------
__global__ void __launch_bounds__(256)
convert_wT(const float* __restrict__ W, int K, int N,
           _Float16* __restrict__ WT, int Kp, int Np,
           long long sW, long long sWT)
{
    const int b = blockIdx.y;
    long long i = (long long)blockIdx.x * 256 + threadIdx.x;
    long long total = (long long)Np * Kp;
    if (i < total) {
        int n = (int)(i / Kp), k = (int)(i % Kp);
        float v = (n < N && k < K) ? W[(long long)b * sW + (long long)k * N + n] : 0.f;
        WT[(long long)b * sWT + i] = (_Float16)v;
    }
}

__global__ void __launch_bounds__(256)
zero_cols_kernel(float* __restrict__ X, int ldx, int col0, int ncol, int rows)
{
    int i = blockIdx.x * 256 + threadIdx.x;
    if (i < rows * ncol) {
        int r = i / ncol, c = i % ncol;
        X[(long long)r * ldx + col0 + c] = 0.f;
    }
}

// ---------------------------------------------------------------------------
// Batched GEMM: C[b,m,n] = act( A[b,m,:] @ W[b,:,n] + bias[b,n] )
// A: f32 [M][lda], lda>=K, K%32==0 (zero-padded). WT: f16 [Np][Kp] transposed.
// Block = 128 thr = 4 waves; tile 64(M) x 16(N), K-step 32.
// Fragments (ISA 7.12.2): A lane L: row L&15, halves {hi*8+e, 16+hi*8+e}
//                         B lane L: col  L&15, halves {hi*8+e, 16+hi*8+e} (lsB is [n][k])
//                         C elem r: N=L&15, M=r+hi*8
// ---------------------------------------------------------------------------
__global__ void __launch_bounds__(128)
gemm_f16_wmma(const float* __restrict__ A, int lda, long long sA,
              const _Float16* __restrict__ WT, int ldw /*Kp*/, long long sW,
              const float* __restrict__ bias, long long sBias,
              float* __restrict__ C, int ldc, long long sC,
              int M, int N, int K /*==Kp*/, int act /*0=none,1=tanh*/)
{
    __shared__ _Float16 lsA[64][32];   // [m][k], rows 64B
    __shared__ _Float16 lsB[16][32];   // [n][k], rows 64B

    const int tid   = threadIdx.x;
    const int wave  = tid >> 5;
    const int lane  = tid & 31;
    const int hlane = lane & 15;
    const int hi    = lane >> 4;
    const int batch = blockIdx.z;
    const int m0    = blockIdx.y * 64;
    const int n0    = blockIdx.x * 16;

    A  += (long long)batch * sA;
    WT += (long long)batch * sW;
    C  += (long long)batch * sC;
    const float* bp = bias ? (bias + (long long)batch * sBias) : nullptr;

    v8f acc = {};
    for (int k0 = 0; k0 < K; k0 += 32) {
        // B tile: 16 rows x 32 halves = 64 x b128, async copy
        if (tid < 64) {
            int r = tid >> 2, c = (tid & 3) * 8;
            async_load_lds_b128(lds_off_of(&lsB[r][c]),
                                WT + (long long)(n0 + r) * ldw + k0 + c);
        }
        // A tile: 64 rows x 32 f32 -> f16, vectorized (float4 -> half4)
        for (int i = tid; i < 512; i += 128) {
            int r = i >> 3, c = (i & 7) * 4;
            f4 v = *reinterpret_cast<const f4*>(A + (long long)(m0 + r) * lda + k0 + c);
            h4 hv = { (_Float16)v[0], (_Float16)v[1], (_Float16)v[2], (_Float16)v[3] };
            *reinterpret_cast<h4*>(&lsA[r][c]) = hv;
        }
        wait_async0();
        __syncthreads();

        V16 af, bf;
        af.h[0] = *reinterpret_cast<const h8*>(&lsA[wave * 16 + hlane][hi * 8]);
        af.h[1] = *reinterpret_cast<const h8*>(&lsA[wave * 16 + hlane][16 + hi * 8]);
        bf.h[0] = *reinterpret_cast<const h8*>(&lsB[hlane][hi * 8]);
        bf.h[1] = *reinterpret_cast<const h8*>(&lsB[hlane][16 + hi * 8]);
        acc = __builtin_amdgcn_wmma_f32_16x16x32_f16(
                  false, af.v, false, bf.v, (short)0, acc, false, false);
        __syncthreads();
    }

    const int nG = n0 + hlane;
    if (nG < N) {
        const float b = bp ? bp[nG] : 0.0f;
        FAcc fa; fa.v = acc;
        #pragma unroll
        for (int r = 0; r < 8; ++r) {
            int mG = m0 + wave * 16 + r + hi * 8;
            float v = fa.f[r] + b;
            if (act == 1) v = tanhf(v);
            C[(long long)mG * ldc + nG] = v;
        }
    }
}

// ---------------------------------------------------------------------------
// BatchNorm (+optional elementwise mul) + optional ReLU. One block per col c.
// In:  X[(c/innerC)*inOuter + c%innerC + r*rowStride] (* mul[r*mulRow + c%innerC])
// Out: Y[(c/innerC)*outOuter + (c%innerC)*outInner + r*outRow]
// ---------------------------------------------------------------------------
__global__ void __launch_bounds__(256)
bn_relu_kernel(const float* __restrict__ X, long long inOuter, int innerC,
               long long rowStride,
               const float* __restrict__ mul, long long mulRow,
               const float* __restrict__ gamma, const float* __restrict__ beta,
               float* __restrict__ Y, long long outOuter, long long outInner,
               long long outRow, int R, int doRelu)
{
    __shared__ float ssum[256], ssq[256];
    __shared__ float sscale, sshift;

    const int c  = blockIdx.x;
    const int co = c / innerC, ci = c % innerC;
    const float* xb = X + (long long)co * inOuter + ci;
    float*       yb = Y + (long long)co * outOuter + (long long)ci * outInner;
    const int t = threadIdx.x;

    float s = 0.f, q = 0.f;
    for (int r = t; r < R; r += 256) {
        float v = xb[(long long)r * rowStride];
        if (mul) v *= mul[(long long)r * mulRow + ci];
        s += v; q += v * v;
    }
    ssum[t] = s; ssq[t] = q;
    __syncthreads();
    for (int off = 128; off > 0; off >>= 1) {
        if (t < off) { ssum[t] += ssum[t + off]; ssq[t] += ssq[t + off]; }
        __syncthreads();
    }
    if (t == 0) {
        float mean = ssum[0] / (float)R;
        float var  = ssq[0] / (float)R - mean * mean;
        float sc   = gamma[c] * rsqrtf(var + 1e-5f);
        sscale = sc;
        sshift = beta[c] - mean * sc;
    }
    __syncthreads();
    const float sc = sscale, sh = sshift;
    for (int r = t; r < R; r += 256) {
        float v = xb[(long long)r * rowStride];
        if (mul) v *= mul[(long long)r * mulRow + ci];
        float y = sc * v + sh;
        if (doRelu) y = fmaxf(y, 0.f);
        yb[(long long)r * outRow] = y;
    }
}

// ---------------------------------------------------------------------------
// Fused gene attention per (pathway p, 64-row batch tile):
//   async per-lane gather genes -> LDS; async b128 copy of WaT[p] tile;
//   cat = [genes(64) | gd(17) | 0-pad ->96];  att = tanh(cat @ Wa + ba)
//   (WMMA 64x96x64); softmax over m; dot = sum_m genes * att.
// ---------------------------------------------------------------------------
#define BT 64
__global__ void __launch_bounds__(128)
gene_attention_kernel(const float* __restrict__ cl,      // [B, G]
                      const int*   __restrict__ gidx,    // [P, 64]
                      const float* __restrict__ gd,      // [P, B, 17]
                      const _Float16* __restrict__ WaT,  // [P][64 n][96 k]
                      const float* __restrict__ ba,      // [P, 64]
                      float* __restrict__ dotOut)        // [P, B]
{
    __shared__ float    geneF[BT][64];
    __shared__ _Float16 catA[BT][96];     // [m-row][k]
    __shared__ _Float16 lsW[64][96];      // [n][k]
    __shared__ float    att[BT][64];
    __shared__ int      idx_s[64];
    __shared__ float    ba_s[64];

    const int p   = blockIdx.y;
    const int b0  = blockIdx.x * BT;
    const int tid = threadIdx.x;
    const int wave = tid >> 5, lane = tid & 31;
    const int hlane = lane & 15, hi = lane >> 4;

    if (tid < 64) {
        idx_s[tid] = gidx[p * 64 + tid];
        ba_s[tid]  = ba[p * 64 + tid];
    }
    __syncthreads();

    // Per-lane async gather of the gene tile (L2-resident cl_features)
    for (int i = tid; i < BT * 64; i += 128) {
        int b = i >> 6, m = i & 63;
        async_load_lds_b32(lds_off_of(&geneF[b][m]),
                           cl + (long long)(b0 + b) * NGENES + idx_s[m]);
    }
    // Async b128 copy of the transposed f16 Wa tile (64x96)
    {
        const h8* wsrc = reinterpret_cast<const h8*>(WaT + (long long)p * (64 * 96));
        h8* wdst = reinterpret_cast<h8*>(&lsW[0][0]);
        for (int i = tid; i < (64 * 96) / 8; i += 128)
            async_load_lds_b128(lds_off_of(wdst + i), wsrc + i);
    }
    // gd columns 64..80 (+ zero pad to 96)
    for (int i = tid; i < BT * 32; i += 128) {
        int b = i >> 5, h = i & 31;
        float v = (h < 17) ? gd[((long long)p * BATCH_N + b0 + b) * 17 + h] : 0.f;
        catA[b][64 + h] = (_Float16)v;
    }
    wait_async0();
    __syncthreads();
    // genes f32 -> f16 into cat
    for (int i = tid; i < BT * 64; i += 128) {
        int b = i >> 6, m = i & 63;
        catA[b][m] = (_Float16)geneF[b][m];
    }
    __syncthreads();

    // 64x96x64 WMMA: each wave owns a 16-row strip, loops 4 N-tiles
    for (int nt = 0; nt < 4; ++nt) {
        v8f acc = {};
        #pragma unroll
        for (int k0 = 0; k0 < 96; k0 += 32) {
            V16 af, bf;
            af.h[0] = *reinterpret_cast<const h8*>(&catA[wave * 16 + hlane][k0 + hi * 8]);
            af.h[1] = *reinterpret_cast<const h8*>(&catA[wave * 16 + hlane][k0 + 16 + hi * 8]);
            bf.h[0] = *reinterpret_cast<const h8*>(&lsW[nt * 16 + hlane][k0 + hi * 8]);
            bf.h[1] = *reinterpret_cast<const h8*>(&lsW[nt * 16 + hlane][k0 + 16 + hi * 8]);
            acc = __builtin_amdgcn_wmma_f32_16x16x32_f16(
                      false, af.v, false, bf.v, (short)0, acc, false, false);
        }
        const int n = nt * 16 + hlane;
        FAcc fa; fa.v = acc;
        #pragma unroll
        for (int r = 0; r < 8; ++r) {
            int mrow = wave * 16 + r + hi * 8;
            att[mrow][n] = tanhf(fa.f[r] + ba_s[n]);
        }
    }
    __syncthreads();

    // Row-wise softmax + weighted dot (tanh-bounded => exp stable)
    if (tid < BT) {
        float sum = 0.f;
        for (int m = 0; m < 64; ++m) {
            float e = __expf(att[tid][m]);
            att[tid][m] = e;
            sum += e;
        }
        float inv = 1.f / sum;
        float d = 0.f;
        for (int m = 0; m < 64; ++m) d += geneF[tid][m] * att[tid][m];
        dotOut[(long long)p * BATCH_N + b0 + tid] = d * inv;
    }
}

// ---------------------------------------------------------------------------
__global__ void __launch_bounds__(128)
final_out_kernel(const float* __restrict__ H, const float* __restrict__ W2,
                 const float* __restrict__ b2, float* __restrict__ out, int R)
{
    int r = blockIdx.x * blockDim.x + threadIdx.x;
    if (r < R) {
        float s = 0.f;
        for (int k = 0; k < 128; ++k) s += H[r * 128 + k] * W2[k];
        out[r] = s + b2[0];
    }
}

// ---------------------------------------------------------------------------
// Host-side helpers
// ---------------------------------------------------------------------------
static inline void launch_conv(hipStream_t st, const float* W, int K, int N,
                               _Float16* WT, int Kp, int Np, int batch)
{
    dim3 g((unsigned)(((long long)Np * Kp + 255) / 256), batch);
    convert_wT<<<g, 256, 0, st>>>(W, K, N, WT, Kp, Np,
                                  (long long)K * N, (long long)Np * Kp);
}

static inline void launch_gemm(hipStream_t st, const float* A, int lda, long long sA,
                               const _Float16* WT, int ldw, long long sW,
                               const float* bias, long long sBias,
                               float* C, int ldc, long long sC,
                               int M, int N, int K, int batch, int act)
{
    dim3 g((N + 15) / 16, M / 64, batch);
    gemm_f16_wmma<<<g, 128, 0, st>>>(A, lda, sA, WT, ldw, sW, bias, sBias,
                                     C, ldc, sC, M, N, K, act);
}

static inline void launch_bn(hipStream_t st, const float* X, long long inOuter,
                             int innerC, long long rowStride,
                             const float* mul, long long mulRow,
                             const float* g, const float* b,
                             float* Y, long long outOuter, long long outInner,
                             long long outRow, int R, int C, int relu)
{
    bn_relu_kernel<<<C, 256, 0, st>>>(X, inOuter, innerC, rowStride, mul, mulRow,
                                      g, b, Y, outOuter, outInner, outRow, R, relu);
}

extern "C" void kernel_launch(void* const* d_in, const int* in_sizes, int n_in,
                              void* d_out, int out_size, void* d_ws, size_t ws_size,
                              hipStream_t stream)
{
    (void)in_sizes; (void)n_in; (void)out_size; (void)ws_size;

    const float* drug  = (const float*)d_in[0];
    const float* cl    = (const float*)d_in[1];
    const int*   gidx  = (const int*)  d_in[2];
    const float* dn_W1 = (const float*)d_in[3];
    const float* dn_b1 = (const float*)d_in[4];
    const float* dn_g1 = (const float*)d_in[5];
    const float* dn_t1 = (const float*)d_in[6];
    const float* dn_W2 = (const float*)d_in[7];
    const float* dn_b2 = (const float*)d_in[8];
    const float* dn_g2 = (const float*)d_in[9];
    const float* dn_t2 = (const float*)d_in[10];
    const float* da_W1 = (const float*)d_in[11];
    const float* da_b1 = (const float*)d_in[12];
    const float* da_g1 = (const float*)d_in[13];
    const float* da_t1 = (const float*)d_in[14];
    const float* da_W2 = (const float*)d_in[15];
    const float* da_b2 = (const float*)d_in[16];
    const float* da_g2 = (const float*)d_in[17];
    const float* da_t2 = (const float*)d_in[18];
    const float* gn_Wd = (const float*)d_in[19];
    const float* gn_bd = (const float*)d_in[20];
    const float* gn_gd = (const float*)d_in[21];
    const float* gn_td = (const float*)d_in[22];
    const float* gn_Wa = (const float*)d_in[23];
    const float* gn_ba = (const float*)d_in[24];
    const float* gn_g1 = (const float*)d_in[25];
    const float* gn_t1 = (const float*)d_in[26];
    const float* pn_Wd = (const float*)d_in[27];
    const float* pn_bd = (const float*)d_in[28];
    const float* pn_gd = (const float*)d_in[29];
    const float* pn_td = (const float*)d_in[30];
    const float* pn_Wa = (const float*)d_in[31];
    const float* pn_ba = (const float*)d_in[32];
    const float* pn_g  = (const float*)d_in[33];
    const float* pn_t  = (const float*)d_in[34];
    const float* out_W1= (const float*)d_in[35];
    const float* out_b1= (const float*)d_in[36];
    const float* out_g1= (const float*)d_in[37];
    const float* out_t1= (const float*)d_in[38];
    const float* out_W2= (const float*)d_in[39];
    const float* out_b2= (const float*)d_in[40];

    const int Bn = 2048;

    // ---- workspace: f32 region then 16B-aligned f16 region ----
    float* ws = (float*)d_ws;
    size_t off = 0;
    auto alloc = [&](size_t n) { float* p = ws + off; off += n; return p; };
    float* Z1     = alloc((size_t)Bn * 256);
    float* Z2     = alloc((size_t)Bn * 128);
    float* Za     = alloc((size_t)Bn * 128);
    float* Datt   = alloc((size_t)Bn * 32);
    float* gd     = alloc((size_t)332 * Bn * 17);
    float* dotRaw = alloc((size_t)332 * Bn);
    float* cat1   = alloc((size_t)Bn * 384);   // [gad(332) | pd(21) | 0-pad]
    float* pdRaw  = alloc((size_t)Bn * 24);
    float* paRaw  = alloc((size_t)Bn * 332);
    float* cat2   = alloc((size_t)Bn * 480);   // [embed(128) | pa(332) | 0-pad]
    float* Hr     = alloc((size_t)Bn * 128);

    _Float16* hb = (_Float16*)(ws + off);
    size_t hoff = 0;
    auto halloc = [&](size_t n) { _Float16* p = hb + hoff; hoff += (n + 7) & ~(size_t)7; return p; };
    _Float16* dn_W1T  = halloc((size_t)256 * 512);
    _Float16* dn_W2T  = halloc((size_t)128 * 256);
    _Float16* da_W1T  = halloc((size_t)128 * 512);
    _Float16* da_W2T  = halloc((size_t)32 * 128);
    _Float16* gn_WdT  = halloc((size_t)332 * 32 * 32);
    _Float16* gn_WaT  = halloc((size_t)332 * 64 * 96);
    _Float16* pn_WdT  = halloc((size_t)32 * 32);
    _Float16* pn_WaT  = halloc((size_t)336 * 384);
    _Float16* out_W1T = halloc((size_t)128 * 480);

    // ---- weight pre-pass: f32 [K][N] -> f16 transposed padded [Np][Kp] ----
    launch_conv(stream, dn_W1, 512, 256, dn_W1T, 512, 256, 1);
    launch_conv(stream, dn_W2, 256, 128, dn_W2T, 256, 128, 1);
    launch_conv(stream, da_W1, 512, 128, da_W1T, 512, 128, 1);
    launch_conv(stream, da_W2, 128, 32,  da_W2T, 128, 32,  1);
    launch_conv(stream, gn_Wd, 32,  17,  gn_WdT, 32,  32,  332);
    launch_conv(stream, gn_Wa, 81,  64,  gn_WaT, 96,  64,  332);
    launch_conv(stream, pn_Wd, 32,  21,  pn_WdT, 32,  32,  1);
    launch_conv(stream, pn_Wa, 353, 332, pn_WaT, 384, 336, 1);
    launch_conv(stream, out_W1, 460, 128, out_W1T, 480, 128, 1);
    // zero the K-padding columns of the concat activations
    zero_cols_kernel<<<(Bn * 31 + 255) / 256, 256, 0, stream>>>(cat1, 384, 353, 31, Bn);
    zero_cols_kernel<<<(Bn * 20 + 255) / 256, 256, 0, stream>>>(cat2, 480, 460, 20, Bn);

    // ---- drug embed MLP ----
    launch_gemm(stream, drug, 512, 0, dn_W1T, 512, 0, dn_b1, 0, Z1, 256, 0,
                Bn, 256, 512, 1, 0);
    launch_bn(stream, Z1, 0, 256, 256, nullptr, 0, dn_g1, dn_t1,
              Z1, 0, 1, 256, Bn, 256, 1);
    launch_gemm(stream, Z1, 256, 0, dn_W2T, 256, 0, dn_b2, 0, Z2, 128, 0,
                Bn, 128, 256, 1, 0);
    launch_bn(stream, Z2, 0, 128, 128, nullptr, 0, dn_g2, dn_t2,
              cat2, 0, 1, 480, Bn, 128, 1);             // -> cat2[:, :128]

    // ---- drug attention MLP ----
    launch_gemm(stream, drug, 512, 0, da_W1T, 512, 0, da_b1, 0, Za, 128, 0,
                Bn, 128, 512, 1, 0);
    launch_bn(stream, Za, 0, 128, 128, nullptr, 0, da_g1, da_t1,
              Za, 0, 1, 128, Bn, 128, 1);
    launch_gemm(stream, Za, 128, 0, da_W2T, 128, 0, da_b2, 0, Datt, 32, 0,
                Bn, 32, 128, 1, 0);
    launch_bn(stream, Datt, 0, 32, 32, nullptr, 0, da_g2, da_t2,
              Datt, 0, 1, 32, Bn, 32, 1);               // drug_att

    // ---- gd = drug_att @ gn_Wd[p] + bd ; BN over b per (p,h) ; relu ----
    launch_gemm(stream, Datt, 32, 0, gn_WdT, 32, 32LL * 32, gn_bd, 17,
                gd, 17, (long long)Bn * 17, Bn, 17, 32, 332, 0);
    launch_bn(stream, gd, (long long)Bn * 17, 17, 17, nullptr, 0, gn_gd, gn_td,
              gd, (long long)Bn * 17, 1, 17, Bn, 332 * 17, 1);

    // ---- fused gather + attention WMMA + tanh + softmax + dot ----
    {
        dim3 ga(Bn / BT, 332);
        gene_attention_kernel<<<ga, 128, 0, stream>>>(cl, gidx, gd, gn_WaT,
                                                      gn_ba, dotRaw);
    }
    // BN over b per p, relu, transpose [P,B] -> cat1[:, :332]
    launch_bn(stream, dotRaw, Bn, 1, 1, nullptr, 0, gn_g1, gn_t1,
              cat1, 1, 0, 384, Bn, 332, 1);

    // ---- pd = relu(BN(drug_att @ pn_Wd)) -> cat1[:, 332:353] ----
    launch_gemm(stream, Datt, 32, 0, pn_WdT, 32, 0, pn_bd, 0, pdRaw, 21, 0,
                Bn, 21, 32, 1, 0);
    launch_bn(stream, pdRaw, 0, 21, 21, nullptr, 0, pn_gd, pn_td,
              cat1 + 332, 0, 1, 384, Bn, 21, 1);

    // ---- pa = relu(BN( tanh(cat1 @ pn_Wa + ba) * gad )) -> cat2[:,128:460] --
    launch_gemm(stream, cat1, 384, 0, pn_WaT, 384, 0, pn_ba, 0, paRaw, 332, 0,
                Bn, 332, 384, 1, 1 /*tanh*/);
    launch_bn(stream, paRaw, 0, 332, 332, cat1, 384, pn_g, pn_t,
              cat2 + 128, 0, 1, 480, Bn, 332, 1);

    // ---- h = relu(BN(cat2 @ out_W1)) ; out = h @ out_W2 + b2 ----
    launch_gemm(stream, cat2, 480, 0, out_W1T, 480, 0, out_b1, 0, Hr, 128, 0,
                Bn, 128, 480, 1, 0);
    launch_bn(stream, Hr, 0, 128, 128, nullptr, 0, out_g1, out_t1,
              Hr, 0, 1, 128, Bn, 128, 1);
    final_out_kernel<<<Bn / 128, 128, 0, stream>>>(Hr, out_W2, out_b2,
                                                   (float*)d_out, Bn);
}
